// ESIM_89300960018823
// MI455X (gfx1250) — matrix operand order repaired and checked by reading, not compiled, not run on hardware
//
#include <hip/hip_runtime.h>
#include <hip/hip_bf16.h>
#include <math.h>

// ---------------------------------------------------------------------------
// ESIM forward for MI455X (gfx1250, wave32, WMMA bf16 -> f32 accumulate)
// ---------------------------------------------------------------------------
#define Tn   128   // T1 == T2
#define Bn   256   // batch
#define En   300   // embed dim
#define Epad 320   // padded to multiple of 32
#define Hn   512   // hidden
#define Gn   2048  // 4*H gates
#define H2n  1024  // 2*H

typedef __attribute__((ext_vector_type(16))) __bf16 v16bf;
typedef __attribute__((ext_vector_type(8)))  float  v8f;
typedef __attribute__((ext_vector_type(4)))  unsigned int u32x4;

union FragU { u32x4 q[2]; v16bf v; };

// A fragment (16x32 bf16, M x K), contiguous K order:
// lane (l&15) = row M; khalf = l>>4 selects K {0..15} vs {16..31}
__device__ __forceinline__ v16bf load_a_frag(const __bf16* __restrict__ row, int khalf) {
    FragU u;
    const __bf16* p = row + khalf * 16;
    u.q[0] = *(const u32x4*)(p);
    u.q[1] = *(const u32x4*)(p + 8);
    return u.v;
}

// B fragment: 16 contiguous bf16 per lane (prepacked fragment-major)
__device__ __forceinline__ v16bf load_b_frag(const __bf16* __restrict__ p) {
    FragU u;
    u.q[0] = *(const u32x4*)(p);
    u.q[1] = *(const u32x4*)(p + 8);
    return u.v;
}

// --------------------------- setup kernels ---------------------------------

// emb[t*B+b, 0:320] = bf16(embed_W[tok[t*B+b], 0:300]), zero padded
__global__ void k_gather_embed(const int* __restrict__ tok,
                               const float* __restrict__ W,
                               __bf16* __restrict__ out) {
    int tb = blockIdx.x;           // T*B
    int c  = threadIdx.x;          // 320
    int id = tok[tb];
    float v = (c < En) ? W[(size_t)id * En + c] : 0.f;
    out[(size_t)tb * Epad + c] = (__bf16)v;
}

// Pack B matrix (= W^T) into per-32x16-tile (K x N), fragment-major bf16.
// B layout per lane: n = (lane&15), k_in_tile = (lane>>4)*16 + elem.
__global__ void k_pack_B(__bf16* __restrict__ dst,
                         const float* __restrict__ Wx, int ldx, int offx, int Kx, int Kxpad,
                         const float* __restrict__ Wh, int ldh, int offh,
                         const float* __restrict__ Wadd, int offadd, float sgn,
                         int Ntiles, int Ktiles, size_t total) {
    size_t idx = (size_t)blockIdx.x * 256 + threadIdx.x;
    if (idx >= total) return;
    int e = (int)(idx & 511);
    size_t tile = idx >> 9;
    int kt = (int)(tile % Ktiles);
    int nt = (int)(tile / Ktiles);
    int lane = e >> 4, p = e & 15;
    int n = nt * 16 + (lane & 15);
    int k = kt * 32 + (lane >> 4) * 16 + p;
    float v;
    if (k < Kxpad) {
        v = (k < Kx) ? Wx[(size_t)n * ldx + offx + k] : 0.f;
        if (Wadd && k < Kx) v += sgn * Wadd[(size_t)n * ldx + offadd + k];
    } else {
        int k2 = k - Kxpad;
        v = Wh ? Wh[(size_t)n * ldh + offh + k2] : 0.f;
    }
    dst[idx] = (__bf16)v;
}

__global__ void k_bias_add(const float* __restrict__ a, const float* __restrict__ b,
                           float* __restrict__ o, int n) {
    int i = blockIdx.x * 256 + threadIdx.x;
    if (i < n) o[i] = a[i] + b[i];
}

__global__ void k_zero2(float* __restrict__ c, __bf16* __restrict__ hbf) {
    int i = blockIdx.x * 256 + threadIdx.x;   // 256*512
    c[i] = 0.f; hbf[i] = (__bf16)0.f;
}

// --------------------------- fused LSTM step --------------------------------
// One launch per time step.  512 cell-chunks (16 Mtiles x 32 Htiles); each block
// owns 2 chunks; its 8 waves compute the 4 gate tiles (i,f,g,o) per chunk with
// WMMA, stage them in LDS, then the block applies the LSTM nonlinearity.
// h is double buffered (read hprev / write hnext) since other blocks still read
// hprev while we write.  c is chunk-private -> in place.
__global__ void k_lstm_step(const __bf16* __restrict__ X, int ldx, int ktX,
                            const __bf16* __restrict__ Hprev,
                            const __bf16* __restrict__ Bp,
                            const float* __restrict__ bias,
                            float* __restrict__ hOut, float* __restrict__ c,
                            __bf16* __restrict__ Hnext, int Ktiles) {
    __shared__ float sg[2 * 4 * 256];          // [chunk][gate][16x16]
    int wave = threadIdx.x >> 5;               // 0..7
    int lane = threadIdx.x & 31;
    int ch   = wave >> 2;                      // chunk in block
    int gate = wave & 3;                       // i,f,g,o
    int chunk = blockIdx.x * 2 + ch;           // 0..511
    int tm = chunk >> 5;                       // batch tile
    int ht = chunk & 31;                       // hidden tile
    int tn = gate * 32 + ht;                   // gate column tile (of 128)
    int khalf = lane >> 4;
    int row = tm * 16 + (lane & 15);

    const __bf16* Xrow = X     + (size_t)row * ldx;
    const __bf16* Hrow = Hprev + (size_t)row * Hn;
    const __bf16* Bt   = Bp + (size_t)tn * Ktiles * 512 + (size_t)lane * 16;

    v8f acc = {};
    int kt = 0;
    for (; kt < ktX; ++kt) {                   // x-part of K
        v16bf a = load_a_frag(Xrow + kt * 32, khalf);
        const __bf16* bp = Bt + (size_t)kt * 512;
        __builtin_prefetch(bp + 512, 0, 1);
        v16bf b = load_b_frag(bp);
        acc = __builtin_amdgcn_wmma_f32_16x16x32_bf16(false, a, false, b,
                                                      (short)0, acc, false, false);
    }
    for (; kt < Ktiles; ++kt) {                // h-part of K
        v16bf a = load_a_frag(Hrow + (kt - ktX) * 32, khalf);
        const __bf16* bp = Bt + (size_t)kt * 512;
        __builtin_prefetch(bp + 512, 0, 1);
        v16bf b = load_b_frag(bp);
        acc = __builtin_amdgcn_wmma_f32_16x16x32_bf16(false, a, false, b,
                                                      (short)0, acc, false, false);
    }

    // stage gate tile (+bias) into LDS
    float bv = bias[tn * 16 + (lane & 15)];
    float* st = sg + (ch * 4 + gate) * 256;
#pragma unroll
    for (int v = 0; v < 8; ++v)
        st[(khalf * 8 + v) * 16 + (lane & 15)] = acc[v] + bv;

    __syncthreads();

    // cell update: 2 chunks x 256 elems, one elem per (thread, chunk)
    int e = threadIdx.x;                       // 0..255 within chunk
    int r = e >> 4, cc = e & 15;
#pragma unroll
    for (int cch = 0; cch < 2; ++cch) {
        int chu = blockIdx.x * 2 + cch;
        int b  = (chu >> 5) * 16 + r;
        int hh = (chu & 31) * 16 + cc;
        const float* g = sg + cch * 4 * 256;
        float gi = g[e], gf = g[256 + e], gg = g[512 + e], go = g[768 + e];
        float si = 1.f / (1.f + expf(-gi));
        float sf = 1.f / (1.f + expf(-gf));
        float so = 1.f / (1.f + expf(-go));
        size_t idx = (size_t)b * Hn + hh;
        float cn = sf * c[idx] + si * tanhf(gg);
        float hn = so * tanhf(cn);
        c[idx] = cn; hOut[idx] = hn; Hnext[idx] = (__bf16)hn;
    }
}

// --------------------------- generic WMMA GEMM -----------------------------

// C[M,N] = act(A[M,K(bf16)] @ Bpacked + bias). One wave per 16x16 tile.
__global__ void k_gemm_bf16(const __bf16* __restrict__ A, int lda,
                            const __bf16* __restrict__ Bp,
                            float* __restrict__ C, int ldc,
                            const float* __restrict__ bias,
                            int Mtiles, int Ntiles, int Ktiles, int act) {
    int wave = blockIdx.x * 8 + (threadIdx.x >> 5);
    int lane = threadIdx.x & 31;
    if (wave >= Mtiles * Ntiles) return;
    int tm = wave / Ntiles, tn = wave % Ntiles;
    int khalf = lane >> 4;
    const __bf16* Arow = A + (size_t)(tm * 16 + (lane & 15)) * lda;
    const __bf16* Bt = Bp + (size_t)tn * Ktiles * 512 + (size_t)lane * 16;
    v8f acc = {};
    for (int kt = 0; kt < Ktiles; ++kt) {
        v16bf a = load_a_frag(Arow + kt * 32, khalf);
        const __bf16* bp = Bt + (size_t)kt * 512;
        __builtin_prefetch(bp + 512, 0, 1);
        v16bf b = load_b_frag(bp);
        acc = __builtin_amdgcn_wmma_f32_16x16x32_bf16(false, a, false, b,
                                                      (short)0, acc, false, false);
    }
    int col = tn * 16 + (lane & 15);
    int r0  = tm * 16 + khalf * 8;
    float bv = bias ? bias[col] : 0.f;
#pragma unroll
    for (int v = 0; v < 8; ++v) {
        float x = acc[v] + bv;
        if (act == 1) x = fmaxf(x, 0.f);
        else if (act == 2) x = tanhf(x);
        C[(size_t)(r0 + v) * ldc + col] = x;
    }
}

// --------------------------- attention & glue ------------------------------

// hcat[b] = concat(hf[b], hr[B-1-b])  (batch-flip of reverse half, per reference)
__global__ void k_assemble_hcat(const float* __restrict__ hf, const float* __restrict__ hr,
                                float* __restrict__ of, __bf16* __restrict__ ob) {
    int idx = blockIdx.x * 256 + threadIdx.x;   // 256*1024
    int b = idx >> 10, d = idx & 1023;
    float v = (d < Hn) ? hf[(size_t)b * Hn + d]
                       : hr[(size_t)(Bn - 1 - b) * Hn + (d - Hn)];
    of[idx] = v; ob[idx] = (__bf16)v;
}

__global__ void k_hadamard_bf(const float* __restrict__ a, const float* __restrict__ bsrc,
                              __bf16* __restrict__ o) {
    int idx = blockIdx.x * 256 + threadIdx.x;   // 256*1024
    o[idx] = (__bf16)(a[idx] * bsrc[idx]);
}

__global__ void k_mask_stats(const float* __restrict__ mask, float* __restrict__ s,
                             float* __restrict__ mx, float* __restrict__ mn) {
    int b = threadIdx.x;   // 256
    float su = 0.f, ma = -1e30f, mi = 1e30f;
    for (int t = 0; t < Tn; ++t) {
        float m = mask[(size_t)t * Bn + b];
        su += m; ma = fmaxf(ma, m); mi = fminf(mi, m);
    }
    s[b] = su; mx[b] = ma; mn[b] = mi;
}

__global__ void k_batch_dot(const float* __restrict__ h1, const float* __restrict__ h2,
                            float* __restrict__ s) {
    int b = threadIdx.x;   // 256
    float acc = 0.f;
    const float* p = h1 + (size_t)b * H2n;
    const float* q = h2 + (size_t)b * H2n;
    for (int d = 0; d < H2n; ++d) acc += p[d] * q[d];
    s[b] = acc;
}

// coef[o,b] = (sum_i mi^2 e_i)/(sum_i mi e_i), e_i = exp(mi*mo*s - max_i(mi*mo*s))
__global__ void k_attn_coef(const float* __restrict__ mask_in,
                            const float* __restrict__ mask_out,
                            const float* __restrict__ s,
                            float* __restrict__ coef) {
    int idx = blockIdx.x * 256 + threadIdx.x;   // 128*256
    int o = idx >> 8, b = idx & 255;
    float c = mask_out[(size_t)o * Bn + b] * s[b];
    float mxw = -1e30f;
    for (int i = 0; i < Tn; ++i)
        mxw = fmaxf(mxw, mask_in[(size_t)i * Bn + b] * c);
    float num = 0.f, den = 0.f;
    for (int i = 0; i < Tn; ++i) {
        float mi = mask_in[(size_t)i * Bn + b];
        float e = expf(mi * c - mxw) * mi;
        den += e; num += mi * e;
    }
    coef[(size_t)o * Bn + b] = num / den;
}

// inp[t,b,h] = relu(m*Pm + cf*Pc + m*cf*Pmc + bias)  -> bf16
__global__ void k_mix(const float* __restrict__ mask, const float* __restrict__ cf,
                      const float* __restrict__ Pm, const float* __restrict__ Pc,
                      const float* __restrict__ Pmc, const float* __restrict__ bias,
                      __bf16* __restrict__ out) {
    size_t idx = (size_t)blockIdx.x * 256 + threadIdx.x;   // T*B*H
    int hh = (int)(idx & 511);
    int tb = (int)(idx >> 9);
    int b = tb & 255;
    float m = mask[tb], a = cf[tb];
    size_t bh = (size_t)b * Hn + hh;
    float x = m * Pm[bh] + a * Pc[bh] + m * a * Pmc[bh] + bias[hh];
    out[idx] = (__bf16)fmaxf(x, 0.f);
}

// logit[b, 0:4096] = [mean(m1*h3), max(m1*h3), mean(m2*h4), max(m2*h4)] -> bf16
__global__ void k_pool(const float* __restrict__ h3, const float* __restrict__ h4,
                       const float* __restrict__ s1, const float* __restrict__ x1,
                       const float* __restrict__ n1,
                       const float* __restrict__ s2, const float* __restrict__ x2,
                       const float* __restrict__ n2,
                       __bf16* __restrict__ out) {
    int idx = blockIdx.x * 256 + threadIdx.x;   // 256*4096
    int b = idx >> 12, d = idx & 4095;
    int chunk = d >> 10, dd = d & 1023;
    const float* h = (chunk < 2) ? h3 : h4;
    float v = h[(size_t)b * H2n + dd];
    float r;
    if ((chunk & 1) == 0) {
        float su = (chunk < 2) ? s1[b] : s2[b];
        r = v * su / su;
    } else {
        float ma = (chunk < 2) ? x1[b] : x2[b];
        float mi = (chunk < 2) ? n1[b] : n2[b];
        r = fmaxf(v * ma, v * mi);
    }
    out[idx] = (__bf16)r;
}

// out[b,:] = softmax(F[b]@op_W^T + op_b)
__global__ void k_op_softmax(const float* __restrict__ F, const float* __restrict__ W,
                             const float* __restrict__ bias, float* __restrict__ out) {
    int b = threadIdx.x;   // 256
    float l[3];
    for (int k = 0; k < 3; ++k) {
        float acc = bias[k];
        const float* f = F + (size_t)b * Hn;
        const float* w = W + (size_t)k * Hn;
        for (int d = 0; d < Hn; ++d) acc += f[d] * w[d];
        l[k] = acc;
    }
    float m = fmaxf(l[0], fmaxf(l[1], l[2]));
    float e0 = expf(l[0] - m), e1 = expf(l[1] - m), e2 = expf(l[2] - m);
    float inv = 1.f / (e0 + e1 + e2);
    out[b * 3 + 0] = e0 * inv; out[b * 3 + 1] = e1 * inv; out[b * 3 + 2] = e2 * inv;
}

// ---------------------------------------------------------------------------

extern "C" void kernel_launch(void* const* d_in, const int* in_sizes, int n_in,
                              void* d_out, int out_size, void* d_ws, size_t ws_size,
                              hipStream_t stream) {
    (void)in_sizes; (void)n_in; (void)out_size; (void)ws_size;
    const int*   x1      = (const int*)  d_in[0];
    const float* x1_mask = (const float*)d_in[1];
    const int*   x2      = (const int*)  d_in[2];
    const float* x2_mask = (const float*)d_in[3];
    const float* embed_W = (const float*)d_in[5];
    const float* Wih[4] = {(const float*)d_in[6],  (const float*)d_in[10],
                           (const float*)d_in[14], (const float*)d_in[18]};
    const float* Whh[4] = {(const float*)d_in[7],  (const float*)d_in[11],
                           (const float*)d_in[15], (const float*)d_in[19]};
    const float* bih[4] = {(const float*)d_in[8],  (const float*)d_in[12],
                           (const float*)d_in[16], (const float*)d_in[20]};
    const float* bhh[4] = {(const float*)d_in[9],  (const float*)d_in[13],
                           (const float*)d_in[17], (const float*)d_in[21]};
    const float* fc1_W = (const float*)d_in[22];
    const float* fc1_b = (const float*)d_in[23];
    const float* fc2_W = (const float*)d_in[24];
    const float* fc2_b = (const float*)d_in[25];
    const float* op_W  = (const float*)d_in[26];
    const float* op_b  = (const float*)d_in[27];
    float* out = (float*)d_out;

    // bump allocator over workspace
    char* base = (char*)d_ws;
    size_t off = 0;
    auto alloc = [&](size_t bytes) -> char* {
        char* p = base + off;
        off += (bytes + 255) & ~(size_t)255;
        return p;
    };
    const size_t TB = (size_t)Tn * Bn;

    __bf16* emb1   = (__bf16*)alloc(TB * Epad * 2);
    __bf16* emb2   = (__bf16*)alloc(TB * Epad * 2);
    __bf16* inp1   = (__bf16*)alloc(TB * Hn * 2);
    __bf16* inp2   = (__bf16*)alloc(TB * Hn * 2);
    const int KTenc = 26, KTdec = 32;           // k-tiles (832 / 1024)
    __bf16* BpE[2] = {(__bf16*)alloc((size_t)KTenc * 128 * 512 * 2),
                      (__bf16*)alloc((size_t)KTenc * 128 * 512 * 2)};
    __bf16* BpD[2] = {(__bf16*)alloc((size_t)KTdec * 128 * 512 * 2),
                      (__bf16*)alloc((size_t)KTdec * 128 * 512 * 2)};
    __bf16* BpU  = (__bf16*)alloc((size_t)32 * 32 * 512 * 2);
    __bf16* BpV  = (__bf16*)alloc((size_t)32 * 32 * 512 * 2);
    __bf16* BpC  = (__bf16*)alloc((size_t)32 * 32 * 512 * 2);
    __bf16* BpF2 = (__bf16*)alloc((size_t)128 * 32 * 512 * 2);
    float* biasL[4];
    for (int i = 0; i < 4; ++i) biasL[i] = (float*)alloc(Gn * 4);
    float*  hfin[8];
    for (int i = 0; i < 8; ++i) hfin[i] = (float*)alloc((size_t)Bn * Hn * 4);
    float*  cbuf  = (float*)alloc((size_t)Bn * Hn * 4);
    __bf16* hbfA  = (__bf16*)alloc((size_t)Bn * Hn * 2);
    __bf16* hbfB  = (__bf16*)alloc((size_t)Bn * Hn * 2);
    float*  hc1 = (float*)alloc((size_t)Bn * H2n * 4);
    float*  hc2 = (float*)alloc((size_t)Bn * H2n * 4);
    float*  hc3 = (float*)alloc((size_t)Bn * H2n * 4);
    float*  hc4 = (float*)alloc((size_t)Bn * H2n * 4);
    __bf16* hc1b = (__bf16*)alloc((size_t)Bn * H2n * 2);
    __bf16* hc2b = (__bf16*)alloc((size_t)Bn * H2n * 2);
    __bf16* hpb  = (__bf16*)alloc((size_t)Bn * H2n * 2);
    float* P1 = (float*)alloc((size_t)Bn * Hn * 4);
    float* P2 = (float*)alloc((size_t)Bn * Hn * 4);
    float* P3 = (float*)alloc((size_t)Bn * Hn * 4);
    float* Q1 = (float*)alloc((size_t)Bn * Hn * 4);
    float* Q2 = (float*)alloc((size_t)Bn * Hn * 4);
    float* sdot = (float*)alloc(Bn * 4);
    float* m1s = (float*)alloc(Bn * 4); float* m1x = (float*)alloc(Bn * 4);
    float* m1n = (float*)alloc(Bn * 4);
    float* m2s = (float*)alloc(Bn * 4); float* m2x = (float*)alloc(Bn * 4);
    float* m2n = (float*)alloc(Bn * 4);
    float* aco = (float*)alloc(TB * 4);   // a[j,b]
    float* bco = (float*)alloc(TB * 4);   // beta~[i,b]
    __bf16* logitA = (__bf16*)alloc((size_t)Bn * 4096 * 2);
    float*  Fbuf   = (float*)alloc((size_t)Bn * Hn * 4);

    // ---- setup -------------------------------------------------------------
    k_gather_embed<<<Tn * Bn, Epad, 0, stream>>>(x1, embed_W, emb1);
    k_gather_embed<<<Tn * Bn, Epad, 0, stream>>>(x2, embed_W, emb2);

    for (int i = 0; i < 2; ++i) {   // encoder: [Wih pad320 ; Whh] K=832 N=2048
        size_t tot = (size_t)KTenc * 128 * 512;
        k_pack_B<<<(int)((tot + 255) / 256), 256, 0, stream>>>(
            BpE[i], Wih[i], En, 0, En, Epad, Whh[i], Hn, 0, nullptr, 0, 0.f,
            128, KTenc, tot);
    }
    for (int i = 0; i < 2; ++i) {   // decoder: [Wih ; Whh] K=1024 N=2048
        size_t tot = (size_t)KTdec * 128 * 512;
        k_pack_B<<<(int)((tot + 255) / 256), 256, 0, stream>>>(
            BpD[i], Wih[2 + i], Hn, 0, Hn, Hn, Whh[2 + i], Hn, 0, nullptr, 0, 0.f,
            128, KTdec, tot);
    }
    {   // fc1 recombined chunks: U=Wa+Wd, V=Wb-Wd, C=Wc  (K=1024,N=512)
        size_t tot = (size_t)32 * 32 * 512;
        int g = (int)((tot + 255) / 256);
        k_pack_B<<<g, 256, 0, stream>>>(BpU, fc1_W, 4096, 0,    H2n, H2n,
                                        nullptr, 0, 0, fc1_W, 3072, 1.f, 32, 32, tot);
        k_pack_B<<<g, 256, 0, stream>>>(BpV, fc1_W, 4096, 1024, H2n, H2n,
                                        nullptr, 0, 0, fc1_W, 3072, -1.f, 32, 32, tot);
        k_pack_B<<<g, 256, 0, stream>>>(BpC, fc1_W, 4096, 2048, H2n, H2n,
                                        nullptr, 0, 0, nullptr, 0, 0.f, 32, 32, tot);
    }
    {   // fc2: K=4096, N=512
        size_t tot = (size_t)128 * 32 * 512;
        k_pack_B<<<(int)((tot + 255) / 256), 256, 0, stream>>>(
            BpF2, fc2_W, 4096, 0, 4096, 4096, nullptr, 0, 0, nullptr, 0, 0.f,
            32, 128, tot);
    }
    for (int i = 0; i < 4; ++i)
        k_bias_add<<<8, 256, 0, stream>>>(bih[i], bhh[i], biasL[i], Gn);

    // ---- LSTM runner (fused step, h double-buffered) -----------------------
    auto run_lstm = [&](const __bf16* X, int ldx, int ktX, int Ktiles,
                        const __bf16* Bp, const float* bias, bool rev, float* hO) {
        k_zero2<<<512, 256, 0, stream>>>(cbuf, hbfA);
        for (int t = 0; t < Tn; ++t) {
            int tt = rev ? (Tn - 1 - t) : t;
            const __bf16* xt = X + (size_t)tt * Bn * ldx;
            __bf16* hr = (t & 1) ? hbfB : hbfA;
            __bf16* hw = (t & 1) ? hbfA : hbfB;
            k_lstm_step<<<256, 256, 0, stream>>>(xt, ldx, ktX, hr, Bp, bias,
                                                 hO, cbuf, hw, Ktiles);
        }
    };

    // encoder: enc/encr applied to both sentences
    run_lstm(emb1, Epad, 10, KTenc, BpE[0], biasL[0], false, hfin[0]);
    run_lstm(emb1, Epad, 10, KTenc, BpE[1], biasL[1], true,  hfin[1]);
    run_lstm(emb2, Epad, 10, KTenc, BpE[0], biasL[0], false, hfin[2]);
    run_lstm(emb2, Epad, 10, KTenc, BpE[1], biasL[1], true,  hfin[3]);

    k_assemble_hcat<<<1024, 256, 0, stream>>>(hfin[0], hfin[1], hc1, hc1b);
    k_assemble_hcat<<<1024, 256, 0, stream>>>(hfin[2], hfin[3], hc2, hc2b);
    k_hadamard_bf<<<1024, 256, 0, stream>>>(hc1, hc2, hpb);
    k_mask_stats<<<1, 256, 0, stream>>>(x1_mask, m1s, m1x, m1n);
    k_mask_stats<<<1, 256, 0, stream>>>(x2_mask, m2s, m2x, m2n);
    k_batch_dot<<<1, 256, 0, stream>>>(hc1, hc2, sdot);
    k_attn_coef<<<128, 256, 0, stream>>>(x1_mask, x2_mask, sdot, aco);  // a[j,b]
    k_attn_coef<<<128, 256, 0, stream>>>(x2_mask, x1_mask, sdot, bco);  // b~[i,b]

    // P/Q projections (M=256, K=1024, N=512): 512 tiles -> 64 blocks
    k_gemm_bf16<<<64, 256, 0, stream>>>(hc1b, H2n, BpU, P1, Hn, nullptr, 16, 32, 32, 0);
    k_gemm_bf16<<<64, 256, 0, stream>>>(hc2b, H2n, BpV, P2, Hn, nullptr, 16, 32, 32, 0);
    k_gemm_bf16<<<64, 256, 0, stream>>>(hpb,  H2n, BpC, P3, Hn, nullptr, 16, 32, 32, 0);
    k_gemm_bf16<<<64, 256, 0, stream>>>(hc2b, H2n, BpU, Q1, Hn, nullptr, 16, 32, 32, 0);
    k_gemm_bf16<<<64, 256, 0, stream>>>(hc1b, H2n, BpV, Q2, Hn, nullptr, 16, 32, 32, 0);

    // fused fc1+relu: inp1[t,b]=relu(m1*P1 + b~*P2 + m1*b~*P3 + b); inp2 analog
    k_mix<<<65536, 256, 0, stream>>>(x1_mask, bco, P1, P2, P3, fc1_b, inp1);
    k_mix<<<65536, 256, 0, stream>>>(x2_mask, aco, Q1, Q2, P3, fc1_b, inp2);

    // decoder
    run_lstm(inp1, Hn, 16, KTdec, BpD[0], biasL[2], false, hfin[4]);
    run_lstm(inp1, Hn, 16, KTdec, BpD[1], biasL[3], true,  hfin[5]);
    run_lstm(inp2, Hn, 16, KTdec, BpD[0], biasL[2], false, hfin[6]);
    run_lstm(inp2, Hn, 16, KTdec, BpD[1], biasL[3], true,  hfin[7]);

    k_assemble_hcat<<<1024, 256, 0, stream>>>(hfin[4], hfin[5], hc3, hc1b);
    k_assemble_hcat<<<1024, 256, 0, stream>>>(hfin[6], hfin[7], hc4, hc2b);

    // pooling -> (B,4096) bf16, fc2 (tanh), op + softmax
    k_pool<<<4096, 256, 0, stream>>>(hc3, hc4, m1s, m1x, m1n, m2s, m2x, m2n, logitA);
    k_gemm_bf16<<<64, 256, 0, stream>>>(logitA, 4096, BpF2, Fbuf, Hn, fc2_b,
                                        16, 32, 128, 2);
    k_op_softmax<<<1, 256, 0, stream>>>(Fbuf, op_W, op_b, out);
}
